// NNclrInfoNCECriterion_13511967113691
// MI455X (gfx1250) — matrix operand order repaired and checked by reading, not compiled, not run on hardware
//
#include <hip/hip_runtime.h>
#include <stdint.h>

// ---------------- problem constants (from reference) ----------------
#define NB    4096      // batch
#define ND    256       // dim
#define NQ    65536     // queue rows
#define NHALF 2048
#define INV_T 10.0f     // 1 / TEMPERATURE

#define QSLICES 32
#define QS_ROWS (NQ / QSLICES)      // 2048 queue rows per slice
#define CSLICES 8
#define CS_COLS (NHALF / CSLICES)   // 256 logit columns per slice

typedef __attribute__((ext_vector_type(16))) __bf16 v16bf;
typedef __attribute__((ext_vector_type(8)))  __bf16 v8bf;
typedef __attribute__((ext_vector_type(8)))  float  v8f;
typedef __attribute__((ext_vector_type(4)))  int    v4i;

// -------- CDNA5 async global->LDS staging (zero data registers) ----------
#if defined(__has_builtin)
#if __has_builtin(__builtin_amdgcn_global_load_async_to_lds_b128)
#define HAVE_ASYNC_LDS 1
#endif
#endif
#ifndef HAVE_ASYNC_LDS
#define HAVE_ASYNC_LDS 0
#endif

// b128 async builtin takes int4-vector pointers: (global v4i*, lds v4i*, imm, imm)
#define GPTR4(p) ((__attribute__((address_space(1))) v4i*)(p))
#define LPTR4(p) ((__attribute__((address_space(3))) v4i*)(p))

__device__ __forceinline__ void wait_asynccnt0() {
#if defined(__has_builtin) && __has_builtin(__builtin_amdgcn_s_wait_asynccnt)
  __builtin_amdgcn_s_wait_asynccnt(0);
#else
  asm volatile("s_wait_asynccnt 0" ::: "memory");
#endif
}

// Stage one 16KB chunk (32 rows x 256 bf16, contiguous) into LDS.
// 256 threads x 4 x 16B. Async path: DMA, no data VGPRs. Fallback: regs.
__device__ __forceinline__ void stage_issue(const uint16_t* __restrict__ gsrc,
                                            uint16_t* ldst, uint4* r) {
  const int t = threadIdx.x;
#if HAVE_ASYNC_LDS
#pragma unroll
  for (int i = 0; i < 4; ++i) {
    __builtin_amdgcn_global_load_async_to_lds_b128(
        GPTR4(gsrc + (size_t)(t + i * 256) * 8),
        LPTR4(ldst + (size_t)(t + i * 256) * 8), 0, 0);
  }
#else
  const uint4* src = (const uint4*)gsrc;
#pragma unroll
  for (int i = 0; i < 4; ++i) r[i] = src[t + i * 256];
#endif
}

__device__ __forceinline__ void stage_commit(uint16_t* ldst, uint4* r) {
#if !HAVE_ASYNC_LDS
  uint4* dst = (uint4*)ldst;
#pragma unroll
  for (int i = 0; i < 4; ++i) dst[threadIdx.x + i * 256] = r[i];
#else
  (void)ldst; (void)r;
#endif
}

__device__ __forceinline__ void stage_drain() {
#if HAVE_ASYNC_LDS
  wait_asynccnt0();
#endif
}

// ---------------- helpers ----------------
__device__ __forceinline__ uint16_t f32_to_bf16(float f) {
  uint32_t u = __float_as_uint(f);
  u += 0x7FFFu + ((u >> 16) & 1u);
  return (uint16_t)(u >> 16);
}

__global__ void cvt_bf16_kernel(const float* __restrict__ src,
                                uint16_t* __restrict__ dst, int n4) {
  int i = blockIdx.x * blockDim.x + threadIdx.x;
  int stride = gridDim.x * blockDim.x;
  for (; i < n4; i += stride) {
    float4 f = ((const float4*)src)[i];
    ushort4 o;
    o.x = f32_to_bf16(f.x); o.y = f32_to_bf16(f.y);
    o.z = f32_to_bf16(f.z); o.w = f32_to_bf16(f.w);
    ((ushort4*)dst)[i] = o;
  }
}

// 16-bit A-fragment, contiguous k order used here (B loads match):
// lanes 0-15 (M=lane):    K = {k0..k0+15}, k0 = s*32
// lanes 16-31 (M=lane-16): same with +16 shift
__device__ __forceinline__ v16bf load_a_frag(const uint16_t* abase, int s, int grp) {
  const int k0 = s * 32 + grp * 16;
  v8bf lo = *(const v8bf*)(abase + k0);
  v8bf hi = *(const v8bf*)(abase + k0 + 8);
  return __builtin_shufflevector(lo, hi, 0, 1, 2, 3, 4, 5, 6, 7,
                                         8, 9, 10, 11, 12, 13, 14, 15);
}

__device__ __forceinline__ v8f wmma_bf16(v16bf a, v16bf b, v8f c) {
  return __builtin_amdgcn_wmma_f32_16x16x32_bf16(false, a, false, b,
                                                 (short)0, c, false, false);
}

// ============================================================================
// Stage 1: partial argmax of embedding @ queue.T over one queue slice.
// grid = (32 rowgroups, 32 slices), 8 waves/block; wave = one 16-row tile.
// B (32 queue rows = 16KB) shared by all 8 waves, double-buffered in LDS,
// filled by async global->LDS DMA overlapped with the WMMAs.
// ============================================================================
__global__ __launch_bounds__(256, 2) void argmax_partial_kernel(
    const uint16_t* __restrict__ ebf,   // [NB x ND] bf16
    const uint16_t* __restrict__ qbf,   // [NQ x ND] bf16
    float* __restrict__ pval,           // [NB x QSLICES]
    int* __restrict__ pidx) {           // [NB x QSLICES]
  __shared__ __align__(32) uint16_t ldsB[2][32 * ND];   // 2 x 16KB

  const int lane  = threadIdx.x & 31;
  const int wave  = threadIdx.x >> 5;
  const int laneN = lane & 15;
  const int grp   = lane >> 4;
  const int row0  = (blockIdx.x * 8 + wave) * 16;
  const int colBase = blockIdx.y * QS_ROWS;

  v16bf a[8];
  {
    const uint16_t* abase = ebf + (size_t)(row0 + laneN) * ND;
#pragma unroll
    for (int s = 0; s < 8; ++s) a[s] = load_a_frag(abase, s, grp);
  }

  float best[8];
  int   bidx[8];
#pragma unroll
  for (int j = 0; j < 8; ++j) { best[j] = -3.4e38f; bidx[j] = 0; }

  const int NCH = QS_ROWS / 32;   // 64 chunks of 32 queue rows
  uint4 r[4];
  stage_issue(qbf + (size_t)colBase * ND, &ldsB[0][0], r);
  stage_commit(&ldsB[0][0], r);
  stage_drain();
  __syncthreads();

  for (int k = 0; k < NCH; ++k) {
    const int cur = k & 1;
    if (k + 1 < NCH)   // start DMA of next chunk before the WMMAs
      stage_issue(qbf + (size_t)(colBase + (k + 1) * 32) * ND, &ldsB[1 - cur][0], r);

    const uint16_t* bb = &ldsB[cur][0];
    v8f c0 = {}, c1 = {};
#pragma unroll
    for (int s = 0; s < 8; ++s) {   // B layout: lane n = col N=n, K contiguous
      v16bf b0 = *(const v16bf*)(bb + laneN * ND + s * 32 + grp * 16);
      v16bf b1 = *(const v16bf*)(bb + (16 + laneN) * ND + s * 32 + grp * 16);
      c0 = wmma_bf16(a[s], b0, c0);
      c1 = wmma_bf16(a[s], b1, c1);
    }
    const int col0 = colBase + k * 32 + laneN;
#pragma unroll
    for (int j = 0; j < 8; ++j) {
      if (c0[j] > best[j]) { best[j] = c0[j]; bidx[j] = col0; }
      if (c1[j] > best[j]) { best[j] = c1[j]; bidx[j] = col0 + 16; }
    }
    if (k + 1 < NCH) stage_commit(&ldsB[1 - cur][0], r);
    stage_drain();
    __syncthreads();
  }

  // reduce (max, argmax) across the 16 N-lanes of each half-wave
#pragma unroll
  for (int j = 0; j < 8; ++j) {
    float bv = best[j];
    int   bi = bidx[j];
#pragma unroll
    for (int st = 1; st < 16; st <<= 1) {
      float ov = __shfl_xor(bv, st, 16);
      int   oi = __shfl_xor(bi, st, 16);
      if (ov > bv || (ov == bv && oi < bi)) { bv = ov; bi = oi; }
    }
    if (laneN == 0) {
      int row = row0 + j + 8 * grp;
      pval[row * QSLICES + blockIdx.y] = bv;
      pidx[row * QSLICES + blockIdx.y] = bi;
    }
  }
}

__global__ void argmax_reduce_kernel(const float* __restrict__ pval,
                                     const int* __restrict__ pidx,
                                     int* __restrict__ idx_out) {
  int row = blockIdx.x * blockDim.x + threadIdx.x;
  if (row < NB) {
    float bv = -3.4e38f; int bi = 0;
#pragma unroll
    for (int s = 0; s < QSLICES; ++s) {
      float v = pval[row * QSLICES + s];
      int   i = pidx[row * QSLICES + s];
      if (v > bv || (v == bv && i < bi)) { bv = v; bi = i; }
    }
    idx_out[row] = bi;
  }
}

// ---------------- gather nearest = queue[idx] (bf16 rows) ----------------
__global__ void gather_kernel(const uint16_t* __restrict__ qbf,
                              const int* __restrict__ idx,
                              uint16_t* __restrict__ nnbf) {
  int t = blockIdx.x * blockDim.x + threadIdx.x;   // NB * 32 threads, 16B each
  int row = t >> 5, part = t & 31;
  if (row < NB) {
    const uint4* src = (const uint4*)(qbf + (size_t)idx[row] * ND);
    uint4*       dst = (uint4*)(nnbf + (size_t)row * ND);
    dst[part] = src[part];
  }
}

__global__ void zero_out_kernel(float* out) {
  if (threadIdx.x == 0 && blockIdx.x == 0) out[0] = 0.0f;
}

// ============================================================================
// Stage 2: partial online-logsumexp CE over one 256-column slice of logits.
// grid = (32 rowgroups, 8 slices), 8 waves/block.
// rowgroups 0..15: logits_ab (B = preds[2048:]); 16..31: logits_ba (B = preds[:2048]).
// ============================================================================
__global__ __launch_bounds__(256, 2) void ce_partial_kernel(
    const uint16_t* __restrict__ nnbf,  // [NB x ND] bf16
    const uint16_t* __restrict__ pbf,   // [NB x ND] bf16
    float* __restrict__ pm, float* __restrict__ ps, float* __restrict__ pd) {
  __shared__ __align__(32) uint16_t ldsB[2][32 * ND];

  const int lane  = threadIdx.x & 31;
  const int wave  = threadIdx.x >> 5;
  const int laneN = lane & 15;
  const int grp   = lane >> 4;
  const int row0  = (blockIdx.x * 8 + wave) * 16;
  const int ba    = (row0 >= NHALF);          // uniform per block (128 | 2048)
  const int boff  = ba ? 0 : NHALF;
  const int rL0   = row0 - (ba ? NHALF : 0);  // local row base == diag col base
  const int colBase = blockIdx.y * CS_COLS;   // local logit-column base

  v16bf a[8];
  {
    const uint16_t* abase = nnbf + (size_t)(row0 + laneN) * ND;
#pragma unroll
    for (int s = 0; s < 8; ++s) a[s] = load_a_frag(abase, s, grp);
  }

  float m[8], ssum[8], dg[8];
#pragma unroll
  for (int j = 0; j < 8; ++j) { m[j] = -3.4e38f; ssum[j] = 0.0f; dg[j] = -3.4e38f; }

  const int NCH = CS_COLS / 32;   // 8 chunks of 32 columns
  uint4 r[4];
  stage_issue(pbf + (size_t)(boff + colBase) * ND, &ldsB[0][0], r);
  stage_commit(&ldsB[0][0], r);
  stage_drain();
  __syncthreads();

  for (int k = 0; k < NCH; ++k) {
    const int cur = k & 1;
    if (k + 1 < NCH)
      stage_issue(pbf + (size_t)(boff + colBase + (k + 1) * 32) * ND, &ldsB[1 - cur][0], r);

    const uint16_t* bb = &ldsB[cur][0];
    v8f c0 = {}, c1 = {};
#pragma unroll
    for (int s = 0; s < 8; ++s) {
      v16bf b0 = *(const v16bf*)(bb + laneN * ND + s * 32 + grp * 16);
      v16bf b1 = *(const v16bf*)(bb + (16 + laneN) * ND + s * 32 + grp * 16);
      c0 = wmma_bf16(a[s], b0, c0);
      c1 = wmma_bf16(a[s], b1, c1);
    }
    const int base0 = colBase + k * 32;      // local col base of subtile 0
    const bool d0 = (base0 == rL0);
    const bool d1 = (base0 + 16 == rL0);
#pragma unroll
    for (int j = 0; j < 8; ++j) {
      float v0 = c0[j] * INV_T;
      float v1 = c1[j] * INV_T;
      if (d0 && laneN == j + 8 * grp) dg[j] = v0;   // logits[i][i]
      if (d1 && laneN == j + 8 * grp) dg[j] = v1;
      float nm = fmaxf(m[j], fmaxf(v0, v1));
      ssum[j] = ssum[j] * __expf(m[j] - nm) + __expf(v0 - nm) + __expf(v1 - nm);
      m[j] = nm;
    }
    if (k + 1 < NCH) stage_commit(&ldsB[1 - cur][0], r);
    stage_drain();
    __syncthreads();
  }

  // combine (m,s) across the 16 N-lanes with rescaling; diag via max
#pragma unroll
  for (int j = 0; j < 8; ++j) {
    float mj = m[j], sj = ssum[j], dj = dg[j];
#pragma unroll
    for (int st = 1; st < 16; st <<= 1) {
      float om = __shfl_xor(mj, st, 16);
      float os = __shfl_xor(sj, st, 16);
      float od = __shfl_xor(dj, st, 16);
      float nm = fmaxf(mj, om);
      sj = sj * __expf(mj - nm) + os * __expf(om - nm);
      mj = nm;
      dj = fmaxf(dj, od);
    }
    if (laneN == 0) {
      int row = row0 + j + 8 * grp;
      pm[row * CSLICES + blockIdx.y] = mj;
      ps[row * CSLICES + blockIdx.y] = sj;
      pd[row * CSLICES + blockIdx.y] = dj;
    }
  }
}

__global__ __launch_bounds__(256) void ce_reduce_kernel(
    const float* __restrict__ pm, const float* __restrict__ ps,
    const float* __restrict__ pd, float* __restrict__ out) {
  __shared__ float red[256];
  int row = blockIdx.x * 256 + threadIdx.x;   // 16 blocks x 256 = 4096 rows
  float m = -3.4e38f, s = 0.0f, d = -3.4e38f;
#pragma unroll
  for (int sl = 0; sl < CSLICES; ++sl) {
    float om = pm[row * CSLICES + sl];
    float os = ps[row * CSLICES + sl];
    float nm = fmaxf(m, om);
    s = s * __expf(m - nm) + os * __expf(om - nm);
    m = nm;
    d = fmaxf(d, pd[row * CSLICES + sl]);
  }
  float rowloss = m + __logf(s) - d;          // logsumexp - diag
  red[threadIdx.x] = rowloss;
  __syncthreads();
  for (int st = 128; st > 0; st >>= 1) {
    if (threadIdx.x < st) red[threadIdx.x] += red[threadIdx.x + st];
    __syncthreads();
  }
  if (threadIdx.x == 0) atomicAdd(out, red[0] * (1.0f / (float)NB));
}

// ---------------- host-side launch ----------------
extern "C" void kernel_launch(void* const* d_in, const int* in_sizes, int n_in,
                              void* d_out, int out_size, void* d_ws, size_t ws_size,
                              hipStream_t stream) {
  const float* emb   = (const float*)d_in[0];   // [4096, 256]
  const float* preds = (const float*)d_in[1];   // [4096, 256]
  const float* queue = (const float*)d_in[2];   // [65536, 256]
  float* out = (float*)d_out;

  // workspace layout (~42 MB)
  uint16_t* qbf  = (uint16_t*)d_ws;                  // NQ*ND bf16
  uint16_t* ebf  = qbf + (size_t)NQ * ND;            // NB*ND
  uint16_t* pbf  = ebf + (size_t)NB * ND;            // NB*ND
  uint16_t* nnbf = pbf + (size_t)NB * ND;            // NB*ND
  int*      idx  = (int*)(nnbf + (size_t)NB * ND);   // NB
  float*    pval = (float*)(idx + NB);               // NB*QSLICES
  int*      pidx = (int*)(pval + (size_t)NB * QSLICES);
  float*    pm   = (float*)(pidx + (size_t)NB * QSLICES);  // NB*CSLICES
  float*    ps   = pm + (size_t)NB * CSLICES;
  float*    pd   = ps + (size_t)NB * CSLICES;

  cvt_bf16_kernel<<<1024, 256, 0, stream>>>(queue, qbf, (NQ * ND) / 4);
  cvt_bf16_kernel<<<256,  256, 0, stream>>>(emb,   ebf, (NB * ND) / 4);
  cvt_bf16_kernel<<<256,  256, 0, stream>>>(preds, pbf, (NB * ND) / 4);

  argmax_partial_kernel<<<dim3(32, QSLICES), 256, 0, stream>>>(ebf, qbf, pval, pidx);
  argmax_reduce_kernel<<<NB / 256, 256, 0, stream>>>(pval, pidx, idx);

  gather_kernel<<<(NB * 32) / 256, 256, 0, stream>>>(qbf, idx, nnbf);

  zero_out_kernel<<<1, 32, 0, stream>>>(out);
  ce_partial_kernel<<<dim3(32, CSLICES), 256, 0, stream>>>(nnbf, pbf, pm, ps, pd);
  ce_reduce_kernel<<<NB / 256, 256, 0, stream>>>(pm, ps, pd, out);
}
